// GQA_34660386079079
// MI455X (gfx1250) — matrix operand order repaired and checked by reading, not compiled, not run on hardware
//
#include <hip/hip_runtime.h>

// Problem constants (match reference)
#define Bc   2
#define Sc   2048
#define Dc   1024
#define Hc   16
#define KVHc 4
#define HDc  64
#define NREP 4
#define S16  (Sc / 16)   // 128 q-tiles per (b,h)

typedef __attribute__((ext_vector_type(16))) __bf16 v16bf;
typedef __attribute__((ext_vector_type(8)))  __bf16 v8bf;
typedef __attribute__((ext_vector_type(8)))  float  v8f;

static __device__ __forceinline__ v8f wmma_bf16(v16bf a, v16bf b, v8f c) {
  // v_wmma_f32_16x16x32_bf16: D = A(16x32) * B(32x16) + C(16x16 f32)
  return __builtin_amdgcn_wmma_f32_16x16x32_bf16(false, a, false, b, (short)0, c,
                                                 false, false);
}

static __device__ __forceinline__ v16bf pack16(v8bf lo, v8bf hi) {
  return __builtin_shufflevector(lo, hi, 0, 1, 2, 3, 4, 5, 6, 7,
                                 8, 9, 10, 11, 12, 13, 14, 15);
}

// ---------------------------------------------------------------------------
// fp32 -> bf16 elementwise convert
__global__ void cvt_f32_bf16(const float* __restrict__ in, __bf16* __restrict__ out,
                             int n) {
  int i = blockIdx.x * blockDim.x + threadIdx.x;
  if (i < n) out[i] = (__bf16)in[i];
}

// fp32 [K x N] -> bf16 transposed [N x K]
__global__ void cvt_transpose_bf16(const float* __restrict__ w, __bf16* __restrict__ wT,
                                   int K, int N) {
  int i = blockIdx.x * blockDim.x + threadIdx.x;
  if (i < K * N) {
    int k = i / N, n = i % N;
    wT[(size_t)n * K + k] = (__bf16)w[i];
  }
}

// ---------------------------------------------------------------------------
// WMMA GEMM: C[M,N] = A[M,K](bf16,row-major) * BT[N,K](bf16)^T + bias
// One wave (32 threads) computes a 16x64 tile. Output scatter by mode:
//   mode 0: Q  -> bf16 [B,H,S,64]
//   mode 1: K  -> bf16 [B,KVH,S,64]
//   mode 2: V  -> bf16 [B,KVH,64,S]  (transposed)
//   mode 3: fp32 row-major [M,N] (final output)
__global__ void gemm_wmma_bf16(const __bf16* __restrict__ A,
                               const __bf16* __restrict__ BT,
                               const float* __restrict__ bias,
                               int N, int K, int mode,
                               __bf16* __restrict__ outb,
                               float* __restrict__ outf) {
  const int lane = threadIdx.x;
  const int half = lane >> 4;      // K-half selector for A/B operands
  const int lj   = lane & 15;      // M (A-rows) / N (B-cols) within tile
  const int ntiles = N / 64;
  const int m0 = (blockIdx.x / ntiles) * 16;
  const int n0 = (blockIdx.x % ntiles) * 64;

  v8f acc[4] = {v8f{}, v8f{}, v8f{}, v8f{}};

  const __bf16* arow = A + (size_t)(m0 + lj) * K;
  for (int k0 = 0; k0 < K; k0 += 32) {
    // A operand: lane=M, 16 contiguous K values starting at 16*half
    const int aoff = k0 + 16 * half;
    v8bf alo = *(const v8bf*)(arow + aoff);
    v8bf ahi = *(const v8bf*)(arow + aoff + 8);
    v16bf a = pack16(alo, ahi);
#pragma unroll
    for (int t = 0; t < 4; ++t) {
      // B operand: lane = column, 16 contiguous K values starting at 16*half
      const __bf16* bp = BT + (size_t)(n0 + 16 * t + lj) * K + k0 + 16 * half;
      v16bf bv = *(const v16bf*)bp;
      acc[t] = wmma_bf16(a, bv, acc[t]);
    }
  }

#pragma unroll
  for (int t = 0; t < 4; ++t) {
    const int col = n0 + 16 * t + lj;
    const float bval = bias[col];
#pragma unroll
    for (int r = 0; r < 8; ++r) {
      const int row = m0 + r + 8 * half;   // C layout: VGPR r -> M = r + 8*(lane/16)
      const float v = acc[t][r] + bval;
      if (mode == 3) {
        outf[(size_t)row * N + col] = v;
      } else {
        const int bb = row / Sc, ss = row % Sc;
        const int hh = col / HDc, dd = col % HDc;
        size_t idx;
        if (mode == 0)      idx = (((size_t)(bb * Hc  + hh)) * Sc + ss) * HDc + dd;
        else if (mode == 1) idx = (((size_t)(bb * KVHc + hh)) * Sc + ss) * HDc + dd;
        else                idx = (((size_t)(bb * KVHc + hh)) * HDc + dd) * Sc + ss;
        outb[idx] = (__bf16)v;
      }
    }
  }
}

// ---------------------------------------------------------------------------
// Flash attention, causal GQA. One wave (32-thread block) per (b, h, 16-row q tile).
// Streams 32-key chunks; scores via 4 WMMA, online softmax in fp32, P bounced
// through LDS to convert C-layout -> A-layout, then 4 WMMA for P*V.
__global__ void attn_flash_wmma(const __bf16* __restrict__ Q,   // [B,H,S,64]
                                const __bf16* __restrict__ Km,  // [B,KVH,S,64]
                                const __bf16* __restrict__ Vt,  // [B,KVH,64,S]
                                __bf16* __restrict__ O) {       // [B,S,D]
  __shared__ __attribute__((aligned(32))) __bf16 P[16 * 32];

  const int lane = threadIdx.x;
  const int half = lane >> 4;
  const int lj   = lane & 15;

  const int bid = blockIdx.x;
  const int qt  = bid % S16;
  const int h   = (bid / S16) % Hc;
  const int b   = bid / (S16 * Hc);
  const int kvh = h / NREP;

  // Q tile as two A operands (d = 0..31 and 32..63), loaded once.
  const __bf16* qrow = Q + (((size_t)(b * Hc + h)) * Sc + qt * 16 + lj) * HDc;
  const int off = 16 * half;
  v16bf aQ0 = pack16(*(const v8bf*)(qrow + off),      *(const v8bf*)(qrow + 8 + off));
  v16bf aQ1 = pack16(*(const v8bf*)(qrow + 32 + off), *(const v8bf*)(qrow + 40 + off));

  const __bf16* Kb = Km + ((size_t)(b * KVHc + kvh)) * Sc * HDc;
  const __bf16* Vb = Vt + ((size_t)(b * KVHc + kvh)) * HDc * Sc;

  v8f acc[4] = {v8f{}, v8f{}, v8f{}, v8f{}};
  float rowM[8], rowL[8];
#pragma unroll
  for (int r = 0; r < 8; ++r) { rowM[r] = -__builtin_inff(); rowL[r] = 0.0f; }

  const int qhi = qt * 16 + 15;
  const float scale = 0.125f;   // 1/sqrt(64)

  for (int kc = 0; kc <= qhi; kc += 32) {
    // ---- scores: S = Q(16x64) @ K^T(64x32), two 16x16 tiles, K-dim split in 2
    const __bf16* kp0 = Kb + (size_t)(kc + lj) * HDc + 16 * half;  // key tile 0
    const __bf16* kp1 = kp0 + 16 * HDc;                            // key tile 1
    if (kc + 32 <= qhi) {
      __builtin_prefetch((const void*)(kp0 + 32 * HDc), 0, 1);     // next K chunk
    }
    v8f s0 = v8f{}, s1 = v8f{};
    s0 = wmma_bf16(aQ0, *(const v16bf*)(kp0),      s0);
    s0 = wmma_bf16(aQ1, *(const v16bf*)(kp0 + 32), s0);
    s1 = wmma_bf16(aQ0, *(const v16bf*)(kp1),      s1);
    s1 = wmma_bf16(aQ1, *(const v16bf*)(kp1 + 32), s1);

    // ---- online softmax (fp32), write P tile (16x32 bf16) to LDS in row-major
    const int j0 = kc + lj;        // global key of score tile 0 column
    const int j1 = kc + 16 + lj;   // global key of score tile 1 column
#pragma unroll
    for (int r = 0; r < 8; ++r) {
      const int m = qt * 16 + r + 8 * half;   // global q row for this VGPR
      float v0 = s0[r] * scale; if (j0 > m) v0 = -__builtin_inff();
      float v1 = s1[r] * scale; if (j1 > m) v1 = -__builtin_inff();
      // row max across the 16 lanes holding this row (xor<16 stays in half)
      float t = fmaxf(v0, v1);
      t = fmaxf(t, __shfl_xor(t, 1, 32));
      t = fmaxf(t, __shfl_xor(t, 2, 32));
      t = fmaxf(t, __shfl_xor(t, 4, 32));
      t = fmaxf(t, __shfl_xor(t, 8, 32));
      const float newM = fmaxf(rowM[r], t);
      const float alpha = __expf(rowM[r] - newM);
      rowM[r] = newM;
      const float p0 = __expf(v0 - newM);
      const float p1 = __expf(v1 - newM);
      const int mr = r + 8 * half;
      P[mr * 32 + lj]      = (__bf16)p0;
      P[mr * 32 + 16 + lj] = (__bf16)p1;
      float rs = p0 + p1;
      rs += __shfl_xor(rs, 1, 32);
      rs += __shfl_xor(rs, 2, 32);
      rs += __shfl_xor(rs, 4, 32);
      rs += __shfl_xor(rs, 8, 32);
      rowL[r] = rowL[r] * alpha + rs;
#pragma unroll
      for (int t4 = 0; t4 < 4; ++t4) acc[t4][r] *= alpha;
    }

    // wave-internal LDS ordering: all P stores visible before A-layout reads
    asm volatile("s_wait_dscnt 0" ::: "memory");

    // ---- P as A operand (16 rows x 32 keys)
    const __bf16* pr = &P[lj * 32];
    v16bf aP = pack16(*(const v8bf*)(pr + off), *(const v8bf*)(pr + 8 + off));

    // ---- O(16x64) += P(16x32) @ V(32x64); V stored transposed -> contiguous keys
#pragma unroll
    for (int t = 0; t < 4; ++t) {
      const __bf16* vp = Vb + (size_t)(16 * t + lj) * Sc + kc + 16 * half;
      acc[t] = wmma_bf16(aP, *(const v16bf*)vp, acc[t]);
    }
    // keep next iteration's P stores from moving above this iteration's P loads
    asm volatile("" ::: "memory");
  }

  // ---- normalize and store O tile: [B,S,D] bf16
#pragma unroll
  for (int t = 0; t < 4; ++t) {
    const int c = h * HDc + 16 * t + lj;
#pragma unroll
    for (int r = 0; r < 8; ++r) {
      const int q = qt * 16 + r + 8 * half;
      O[((size_t)b * Sc + q) * Dc + c] = (__bf16)(acc[t][r] * (1.0f / rowL[r]));
    }
  }
}

// ---------------------------------------------------------------------------
extern "C" void kernel_launch(void* const* d_in, const int* in_sizes, int n_in,
                              void* d_out, int out_size, void* d_ws, size_t ws_size,
                              hipStream_t stream) {
  // inputs: x, mask, wq, bq, wk, bk, wv, bv, wo, bo  (mask unused: causal analytic)
  const float* x  = (const float*)d_in[0];
  const float* wq = (const float*)d_in[2];
  const float* bq = (const float*)d_in[3];
  const float* wk = (const float*)d_in[4];
  const float* bk = (const float*)d_in[5];
  const float* wv = (const float*)d_in[6];
  const float* bv = (const float*)d_in[7];
  const float* wo = (const float*)d_in[8];
  const float* bo = (const float*)d_in[9];
  float* out = (float*)d_out;

  const size_t BS = (size_t)Bc * Sc;                 // 4096
  const size_t nX = BS * Dc;                         // 4,194,304
  const size_t nKV = (size_t)Bc * KVHc * Sc * HDc;   // 1,048,576

  size_t off = 0;
  auto take = [&](size_t elems) {
    void* p = (char*)d_ws + off;
    off += (elems * sizeof(__bf16) + 255) & ~(size_t)255;
    return (__bf16*)p;
  };
  __bf16* xbf  = take(nX);
  __bf16* wqT  = take((size_t)Dc * Dc);
  __bf16* wkT  = take((size_t)KVHc * HDc * Dc);
  __bf16* wvT  = take((size_t)KVHc * HDc * Dc);
  __bf16* woT  = take((size_t)Dc * Dc);
  __bf16* Qbf  = take(nX);
  __bf16* Kbf  = take(nKV);
  __bf16* Vtbf = take(nKV);
  __bf16* Obf  = take(nX);

  // 1) precision conversion / weight transposes
  cvt_f32_bf16<<<(int)((nX + 255) / 256), 256, 0, stream>>>(x, xbf, (int)nX);
  cvt_transpose_bf16<<<(Dc * Dc + 255) / 256, 256, 0, stream>>>(wq, wqT, Dc, Dc);
  cvt_transpose_bf16<<<(Dc * KVHc * HDc + 255) / 256, 256, 0, stream>>>(wk, wkT, Dc, KVHc * HDc);
  cvt_transpose_bf16<<<(Dc * KVHc * HDc + 255) / 256, 256, 0, stream>>>(wv, wvT, Dc, KVHc * HDc);
  cvt_transpose_bf16<<<(Dc * Dc + 255) / 256, 256, 0, stream>>>(wo, woT, Dc, Dc);

  // 2) QKV projections (one wave per 16x64 tile)
  const int gQ = (int)(BS / 16) * (Dc / 64);              // 4096
  const int gK = (int)(BS / 16) * ((KVHc * HDc) / 64);    // 1024
  gemm_wmma_bf16<<<gQ, 32, 0, stream>>>(xbf, wqT, bq, Dc,          Dc, 0, Qbf,  nullptr);
  gemm_wmma_bf16<<<gK, 32, 0, stream>>>(xbf, wkT, bk, KVHc * HDc,  Dc, 1, Kbf,  nullptr);
  gemm_wmma_bf16<<<gK, 32, 0, stream>>>(xbf, wvT, bv, KVHc * HDc,  Dc, 2, Vtbf, nullptr);

  // 3) causal GQA flash attention: one wave per (b,h,qtile)
  attn_flash_wmma<<<Bc * Hc * S16, 32, 0, stream>>>(Qbf, Kbf, Vtbf, Obf);

  // 4) output projection -> fp32 d_out
  gemm_wmma_bf16<<<gQ, 32, 0, stream>>>(Obf, woT, bo, Dc, Dc, 3, nullptr, out);
}
